// AsyncNaiveLinguistic_1297080124021
// MI455X (gfx1250) — matrix operand order repaired and checked by reading, not compiled, not run on hardware
//
#include <hip/hip_runtime.h>

// ---------------------------------------------------------------------------
// Problem constants (from reference): B=128, T=1024, D=300, H=128, 4H=512
// ---------------------------------------------------------------------------
#define Bsz 128
#define Tsz 1024
#define Dsz 300
#define Hsz 128
#define Gsz 512                  // 4*H
#define BT  (Bsz * Tsz)          // 131072

typedef __attribute__((ext_vector_type(16))) __bf16 v16bf;
typedef __attribute__((ext_vector_type(8)))  __bf16 v8bf;
typedef __attribute__((ext_vector_type(8)))  float  v8f;
typedef __attribute__((ext_vector_type(4)))  float  f4;
typedef __attribute__((ext_vector_type(4)))  int    v4i;

// CDNA5 async global->LDS path (ASYNCcnt), guarded for toolchain support.
#if defined(__HIP_DEVICE_COMPILE__) && __has_builtin(__builtin_amdgcn_global_load_async_to_lds_b128)
#define HAVE_ASYNC_LDS 1
#else
#define HAVE_ASYNC_LDS 0
#endif

#if HAVE_ASYNC_LDS
static __device__ __forceinline__ void async_copy_b128(const void* gsrc, void* ldst) {
    // Builtin signature (from hipcc diagnostic): (v4i AS1*, v4i AS3*, Ii, Ii).
    // Flat LDS address truncates to the LDS offset (aperture in high bits).
    __builtin_amdgcn_global_load_async_to_lds_b128(
        (__attribute__((address_space(1))) v4i*)(uintptr_t)gsrc,
        (__attribute__((address_space(3))) v4i*)(unsigned int)(uintptr_t)ldst,
        /*offset=*/0, /*cpol=*/0);
}
static __device__ __forceinline__ void wait_async0() {
#if __has_builtin(__builtin_amdgcn_s_wait_asynccnt)
    __builtin_amdgcn_s_wait_asynccnt(0);
    __asm__ __volatile__("" ::: "memory");
#else
    __asm__ __volatile__("s_wait_asynccnt 0x0" ::: "memory");
#endif
}
#endif

static __device__ __forceinline__ float fast_sigmoid(float x) {
    float e = __expf(-x);
    return 1.0f / (1.0f + e);
}
static __device__ __forceinline__ float fast_tanh(float x) {
    x = fminf(15.0f, fmaxf(-15.0f, x));
    float e = __expf(2.0f * x);
    return (e - 1.0f) / (e + 1.0f);
}

static __device__ __forceinline__ v8f wmma_bf16(v16bf a, v16bf b, v8f c) {
    return __builtin_amdgcn_wmma_f32_16x16x32_bf16(
        /*neg_a=*/false, a, /*neg_b=*/false, b,
        /*c_mod=*/(short)0, c, /*reuse_a=*/false, /*reuse_b=*/false);
}

// ---------------------------------------------------------------------------
// Kernel 1: P[t][b][g] = sum_d x[b,t,d] * w_ih[g,d] + b_ih[g]   (bf16 out)
// One wave per 16(M = flat b*T+t) x 16(N = gate) tile, K = 300 (9 full
// 32-chunks + zero-padded tail).
// ---------------------------------------------------------------------------
__global__ __launch_bounds__(256) void proj_kernel(
    const float* __restrict__ x, const float* __restrict__ w_ih,
    const float* __restrict__ b_ih, __bf16* __restrict__ P)
{
    const int wave = (blockIdx.x * blockDim.x + threadIdx.x) >> 5;
    const int lane = threadIdx.x & 31;
    const int half = lane >> 4;
    const int l    = lane & 15;
    const int mt   = wave >> 5;        // 8192 M-tiles
    const int nt   = wave & 31;        // 32 N-tiles

    const float* xrow = x    + (size_t)(mt * 16 + l) * Dsz;  // A row (M = l)
    const float* wrow = w_ih + (size_t)(nt * 16 + l) * Dsz;  // B col (N = l)

    v8f acc = {};
#pragma unroll
    for (int kc = 0; kc < 9; ++kc) {
        const int k0 = kc * 32;
        // A fragment: half 0 -> K 0..15, half 1 -> K 16..31 (contiguous d)
        const f4* xa = (const f4*)(xrow + k0 + half * 16);
        const f4* xb = (const f4*)(xrow + k0 + half * 16 + 8);
        f4 a0 = xa[0], a1 = xa[1], a2 = xb[0], a3 = xb[1];
        // B fragment: half 0 -> K 0..15, half 1 -> K 16..31 (contiguous d)
        const f4* wb = (const f4*)(wrow + k0 + half * 16);
        f4 b0 = wb[0], b1 = wb[1], b2 = wb[2], b3 = wb[3];
        v16bf a, b;
#pragma unroll
        for (int e = 0; e < 4; ++e) {
            a[e]      = (__bf16)a0[e];  a[4 + e]  = (__bf16)a1[e];
            a[8 + e]  = (__bf16)a2[e];  a[12 + e] = (__bf16)a3[e];
            b[e]      = (__bf16)b0[e];  b[4 + e]  = (__bf16)b1[e];
            b[8 + e]  = (__bf16)b2[e];  b[12 + e] = (__bf16)b3[e];
        }
        acc = wmma_bf16(a, b, acc);
    }
    {   // tail chunk: k0 = 288, valid k < 300, zero pad the rest
        const int k0 = 288;
        v16bf a = {}, b = {};
#pragma unroll
        for (int e = 0; e < 8; ++e) {
            int ka = k0 + half * 16 + e;
            int kb = k0 + half * 16 + 8 + e;
            a[e]     = (ka < Dsz) ? (__bf16)xrow[ka] : (__bf16)0.0f;
            a[8 + e] = (kb < Dsz) ? (__bf16)xrow[kb] : (__bf16)0.0f;
        }
#pragma unroll
        for (int e = 0; e < 16; ++e) {
            int k = k0 + half * 16 + e;
            b[e] = (k < Dsz) ? (__bf16)wrow[k] : (__bf16)0.0f;
        }
        acc = wmma_bf16(a, b, acc);
    }

    const int col  = nt * 16 + l;          // gate index (C: N = l)
    const float bi = b_ih[col];
#pragma unroll
    for (int r = 0; r < 8; ++r) {
        int flat = mt * 16 + half * 8 + r; // flat b*T + t  (C: M = half*8+r)
        int bb = flat >> 10;               // / 1024
        int tt = flat & 1023;
        P[(size_t)(tt * Bsz + bb) * Gsz + col] = (__bf16)(acc[r] + bi);
    }
}

// ---------------------------------------------------------------------------
// Kernel 2: persistent LSTM recurrence. 8 workgroups x 16 batch rows.
// w_hh^T kept in registers as pre-swizzled bf16 WMMA B-fragments.
// h (bf16) / c (f32) / gate activations live in LDS; 2 barriers per step.
// The next step's x-projection slice is staged into LDS with
// GLOBAL_LOAD_ASYNC_TO_LDS_B128 (ASYNCcnt), overlapped with compute.
// Each wave copies exactly the 2KB region it alone reads -> wave-local wait.
// ---------------------------------------------------------------------------
__global__ __launch_bounds__(256, 1) void lstm_kernel(
    const __bf16* __restrict__ P,     // [T][B][4H] bf16
    const float* __restrict__ w_hh,   // [4H][H]
    const float* __restrict__ b_hh,   // [4H]
    __bf16* __restrict__ hs)          // [B][T][H] bf16 out
{
    __shared__ __align__(16) __bf16 hbuf[16][144];   // stride 288B (16B mult)
    __shared__ __align__(16) float cbuf[16][128];
    __shared__ __align__(16) float gbuf[16][512];
#if HAVE_ASYNC_LDS
    __shared__ __align__(16) __bf16 pbuf[16][512];   // staged P slice (16KB)
#endif

    const int tid  = threadIdx.x;
    const int w    = tid >> 5;         // wave 0..7 -> N-tiles w*4..w*4+3
    const int lane = tid & 31;
    const int half = lane >> 4;
    const int l    = lane & 15;
    const int b0   = blockIdx.x * 16;

    for (int i = tid; i < 16 * 144; i += 256) ((__bf16*)hbuf)[i] = (__bf16)0.0f;
    for (int i = tid; i < 16 * 128; i += 256) ((float*)cbuf)[i]  = 0.0f;

    // Preload w_hh^T B-fragments (bf16) into registers: 4 N-tiles x 4 K-chunks.
    v16bf Bf[4][4];
    float bhh[4];
#pragma unroll
    for (int j = 0; j < 4; ++j) {
        const int g = (w * 4 + j) * 16 + l;      // B: N = l
        bhh[j] = b_hh[g];
#pragma unroll
        for (int kc = 0; kc < 4; ++kc) {
            const f4* p = (const f4*)(w_hh + (size_t)g * Hsz + kc * 32 + half * 16);
            f4 q0 = p[0], q1 = p[1], q2 = p[2], q3 = p[3];
#pragma unroll
            for (int e = 0; e < 4; ++e) {
                Bf[j][kc][e]      = (__bf16)q0[e];
                Bf[j][kc][4 + e]  = (__bf16)q1[e];
                Bf[j][kc][8 + e]  = (__bf16)q2[e];
                Bf[j][kc][12 + e] = (__bf16)q3[e];
            }
        }
    }
    const int gateType = w >> 1;   // 0=i, 1=f, 2=g(cand), 3=o

#if HAVE_ASYNC_LDS
    // Stage step t's P slice for this wave: 16 rows x 64 gates (2KB) as
    // 128 x 16B segments spread over 32 lanes x 4 ops.
    auto issueP = [&](int tt) {
        const __bf16* Ps = P + (size_t)tt * Bsz * Gsz;
#pragma unroll
        for (int q = 0; q < 4; ++q) {
            const int s   = q * 32 + lane;
            const int m   = s >> 3;
            const int seg = s & 7;
            async_copy_b128(Ps + (size_t)(b0 + m) * Gsz + w * 64 + seg * 8,
                            &pbuf[m][w * 64 + seg * 8]);
        }
    };
    issueP(0);
#endif
    __syncthreads();

    for (int t = 0; t < Tsz; ++t) {
        const __bf16* Pt = P + (size_t)t * Bsz * Gsz;
#if !HAVE_ASYNC_LDS
        if (t + 1 < Tsz)   // pull next step's 16KB slice toward the caches
            __builtin_prefetch(P + (size_t)(t + 1) * Bsz * Gsz +
                               (size_t)b0 * Gsz + (size_t)tid * 32, 0, 1);
#endif

        v8f acc[4] = {v8f{}, v8f{}, v8f{}, v8f{}};
#pragma unroll
        for (int kc = 0; kc < 4; ++kc) {
            const v8bf* hrow = (const v8bf*)&hbuf[l][0];
            v8bf alo = hrow[kc * 4 + half * 2];
            v8bf ahi = hrow[kc * 4 + half * 2 + 1];
            v16bf a = __builtin_shufflevector(alo, ahi,
                0, 1, 2, 3, 4, 5, 6, 7, 8, 9, 10, 11, 12, 13, 14, 15);
#pragma unroll
            for (int j = 0; j < 4; ++j)
                acc[j] = wmma_bf16(a, Bf[j][kc], acc[j]);
        }

#if HAVE_ASYNC_LDS
        wait_async0();   // this wave's staged P slice is ready in LDS
#endif
        // epilogue: + x-projection + b_hh, activate, scatter into gate buffer
#pragma unroll
        for (int j = 0; j < 4; ++j) {
            const int g = (w * 4 + j) * 16 + l;
#pragma unroll
            for (int r = 0; r < 8; ++r) {
                const int m = half * 8 + r;    // batch-local row
#if HAVE_ASYNC_LDS
                float xp = (float)pbuf[m][g];
#else
                float xp = (float)Pt[(size_t)(b0 + m) * Gsz + g];
#endif
                float pre = acc[j][r] + xp + bhh[j];
                float act = (gateType == 2) ? fast_tanh(pre) : fast_sigmoid(pre);
                gbuf[m][g] = act;
            }
        }
        __syncthreads();

#if HAVE_ASYNC_LDS
        if (t + 1 < Tsz) issueP(t + 1);   // overlap with update + next WMMA
#endif
        // c/h update: 16 rows x 128 cols, 8 consecutive cols per thread
        {
            const int rowU = tid >> 4;
            const int c0   = (tid & 15) * 8;
            const f4* grow = (const f4*)&gbuf[rowU][0];
            f4 iv0 = grow[c0 / 4],           iv1 = grow[c0 / 4 + 1];
            f4 fv0 = grow[(128 + c0) / 4],   fv1 = grow[(128 + c0) / 4 + 1];
            f4 gv0 = grow[(256 + c0) / 4],   gv1 = grow[(256 + c0) / 4 + 1];
            f4 ov0 = grow[(384 + c0) / 4],   ov1 = grow[(384 + c0) / 4 + 1];
            f4* crow = (f4*)&cbuf[rowU][c0];
            f4 cv0 = crow[0], cv1 = crow[1];
            f4 cn0, cn1;
            v8bf hv;
#pragma unroll
            for (int u = 0; u < 4; ++u) {
                cn0[u] = fv0[u] * cv0[u] + iv0[u] * gv0[u];
                cn1[u] = fv1[u] * cv1[u] + iv1[u] * gv1[u];
                hv[u]     = (__bf16)(ov0[u] * fast_tanh(cn0[u]));
                hv[4 + u] = (__bf16)(ov1[u] * fast_tanh(cn1[u]));
            }
            crow[0] = cn0;
            crow[1] = cn1;
            *(v8bf*)&hbuf[rowU][c0] = hv;
            *(v8bf*)&hs[(size_t)((b0 + rowU) * Tsz + t) * Hsz + c0] = hv;
        }
        __syncthreads();
    }
}

// ---------------------------------------------------------------------------
// Kernel 3: out[b,t] = ((hs @ w1^T + b1) @ w2^T + b2) * mask.
// One wave per 16-row tile; 4 N-tiles (64 mid units) x 4 K-chunks WMMA,
// then a width-16 shuffle reduction against w2.
// ---------------------------------------------------------------------------
__global__ __launch_bounds__(256) void head_kernel(
    const __bf16* __restrict__ hs, const float* __restrict__ w1,
    const float* __restrict__ b1, const float* __restrict__ w2,
    const float* __restrict__ b2, const float* __restrict__ mask,
    float* __restrict__ out)
{
    const int wave = (blockIdx.x * blockDim.x + threadIdx.x) >> 5; // 8192 tiles
    const int lane = threadIdx.x & 31;
    const int half = lane >> 4;
    const int l    = lane & 15;
    const int mt   = wave;

    // B fragments from w1 [64][128]
    v16bf Bf[4][4];
    float w2l[4], b1l[4];
#pragma unroll
    for (int j = 0; j < 4; ++j) {
        const int col = j * 16 + l;
        w2l[j] = w2[col];
        b1l[j] = b1[col];
#pragma unroll
        for (int kc = 0; kc < 4; ++kc) {
            const f4* p = (const f4*)(w1 + (size_t)col * Hsz + kc * 32 + half * 16);
            f4 q0 = p[0], q1 = p[1], q2 = p[2], q3 = p[3];
#pragma unroll
            for (int e = 0; e < 4; ++e) {
                Bf[j][kc][e]      = (__bf16)q0[e];
                Bf[j][kc][4 + e]  = (__bf16)q1[e];
                Bf[j][kc][8 + e]  = (__bf16)q2[e];
                Bf[j][kc][12 + e] = (__bf16)q3[e];
            }
        }
    }

    v8f acc[4] = {v8f{}, v8f{}, v8f{}, v8f{}};
    const v8bf* hrow = (const v8bf*)(hs + (size_t)(mt * 16 + l) * Hsz);
#pragma unroll
    for (int kc = 0; kc < 4; ++kc) {
        v8bf alo = hrow[kc * 4 + half * 2];
        v8bf ahi = hrow[kc * 4 + half * 2 + 1];
        v16bf a = __builtin_shufflevector(alo, ahi,
            0, 1, 2, 3, 4, 5, 6, 7, 8, 9, 10, 11, 12, 13, 14, 15);
#pragma unroll
        for (int j = 0; j < 4; ++j)
            acc[j] = wmma_bf16(a, Bf[j][kc], acc[j]);
    }

    const float bias2 = b2[0];
#pragma unroll
    for (int r = 0; r < 8; ++r) {
        float p = 0.0f;
#pragma unroll
        for (int j = 0; j < 4; ++j)
            p += (acc[j][r] + b1l[j]) * w2l[j];
        // reduce across the 16 lanes of this half-wave (rows differ per half)
        p += __shfl_xor(p, 1, 32);
        p += __shfl_xor(p, 2, 32);
        p += __shfl_xor(p, 4, 32);
        p += __shfl_xor(p, 8, 32);
        if (l == 0) {
            const int m = mt * 16 + half * 8 + r;   // flat b*T + t
            out[m] = (p + bias2) * mask[m];
        }
    }
}

// ---------------------------------------------------------------------------
// Host launcher. Inputs (setup_inputs order):
// 0 x, 1 seq_length, 2 lstm_masks, 3 w_ih, 4 w_hh, 5 b_ih, 6 b_hh,
// 7 w1, 8 b1, 9 w2, 10 b2.
// Workspace: P bf16 [T][B][512] (128 MB) then hs bf16 [B][T][128] (32 MB).
// ---------------------------------------------------------------------------
extern "C" void kernel_launch(void* const* d_in, const int* in_sizes, int n_in,
                              void* d_out, int out_size, void* d_ws, size_t ws_size,
                              hipStream_t stream)
{
    (void)in_sizes; (void)n_in; (void)out_size; (void)ws_size;
    const float* x    = (const float*)d_in[0];
    const float* mask = (const float*)d_in[2];
    const float* w_ih = (const float*)d_in[3];
    const float* w_hh = (const float*)d_in[4];
    const float* b_ih = (const float*)d_in[5];
    const float* b_hh = (const float*)d_in[6];
    const float* w1   = (const float*)d_in[7];
    const float* b1   = (const float*)d_in[8];
    const float* w2   = (const float*)d_in[9];
    const float* b2   = (const float*)d_in[10];
    float* out = (float*)d_out;

    __bf16* P  = (__bf16*)d_ws;
    __bf16* hs = (__bf16*)((char*)d_ws + (size_t)BT * Gsz * sizeof(__bf16));

    // 8192 M-tiles x 32 N-tiles = 262144 waves, 8 waves/block
    proj_kernel<<<32768, 256, 0, stream>>>(x, w_ih, b_ih, P);
    // persistent recurrence: one WG per 16 batch rows
    lstm_kernel<<<Bsz / 16, 256, 0, stream>>>(P, w_hh, b_hh, hs);
    // 8192 tiles, 8 waves/block
    head_kernel<<<BT / 16 / 8, 256, 0, stream>>>(hs, w1, b1, w2, b2, mask, out);
}
